// StateSpaceModelAttentionWithSSD_37065567764790
// MI455X (gfx1250) — matrix operand order repaired and checked by reading, not compiled, not run on hardware
//
#include <hip/hip_runtime.h>

// B=256, S=256 -> M = 65536 rows; D = 448; 14 diagonal 32x32 blocks.
#define DM 448
#define MROWS 65536

typedef __attribute__((ext_vector_type(16))) __bf16       v16bf;
typedef __attribute__((ext_vector_type(8)))  float        v8f;
typedef __attribute__((ext_vector_type(4))) unsigned int  u32x4;
typedef __attribute__((ext_vector_type(8)))  int          i32x8;
typedef __attribute__((ext_vector_type(4)))  int          i32x4;

union FragU { v16bf v; u32x4 q[2]; };

__device__ inline v8f v8f_zero() {
  v8f z = {0.f, 0.f, 0.f, 0.f, 0.f, 0.f, 0.f, 0.f};
  return z;
}

// A-matrix fragment (16x32 bf16): lane<16 holds K={base..base+7,16+base..}, base=0;
// lane>=16: base=8.  Two 16-byte loads.
__device__ inline v16bf frag_a(const __bf16* row, int half) {
  FragU f;
  const int b = half * 8;
  f.q[0] = *(const u32x4*)(row + b);
  f.q[1] = *(const u32x4*)(row + b + 16);
  return f.v;
}

// B-matrix fragment (32x16 bf16), B staged as [N][K]: lane holds N=lane&15,
// K = {base..base+7, 16+base..16+base+7}, base = 8*half.  Two 16-byte loads.
__device__ inline v16bf frag_b(const __bf16* row, int half) {
  FragU f;
  const int b = half * 8;
  f.q[0] = *(const u32x4*)(row + b);
  f.q[1] = *(const u32x4*)(row + b + 16);
  return f.v;
}

__device__ inline v8f wmma_bf16(v16bf a, v16bf b, v8f c) {
  // (neg_a, A, neg_b, B, c_mod, C, reuse_a, reuse_b)
  return __builtin_amdgcn_wmma_f32_16x16x32_bf16(false, a, false, b, (short)0, c,
                                                 false, false);
}

__device__ inline float frag_sum(v16bf f) {
  float s = 0.f;
#pragma unroll
  for (int i = 0; i < 16; ++i) s += (float)f[i];
  return s;
}

// Low 32 bits of a flat pointer into the LDS aperture = byte offset within the
// workgroup's LDS allocation (ISA 10.2).
__device__ inline unsigned lds_off(const void* p) {
  return (unsigned)(unsigned long long)p;
}

// ---------------------------------------------------------------------------
// Tensor Data Mover: 2D tile (global bf16 -> LDS), row length tile_d0 elems,
// tile_d1 rows, line stride `stride0` elems.  Pads +16B after every 64B row
// (pad_interval code 3 = 16 DWORDs, pad_amount code 3 = 4 DWORDs) so the LDS
// image has 40-element rows, matching frag_a/frag_b addressing.
// D# packing per CDNA5 ISA (data_size=1 -> 2B).
// ---------------------------------------------------------------------------
__device__ inline void tdm_load_2d(unsigned lds_addr, const void* gaddr,
                                   unsigned tensor_d0, unsigned tensor_d1,
                                   unsigned tile_d0, unsigned tile_d1,
                                   unsigned stride0) {
  unsigned long long ga = (unsigned long long)gaddr;
  u32x4 g0;
  g0[0] = 1u;                                   // count=1, user descriptor
  g0[1] = lds_addr;                             // lds_addr [63:32]
  g0[2] = (unsigned)(ga & 0xffffffffu);         // global_addr lo
  g0[3] = (unsigned)((ga >> 32) & 0x01ffffffu)  // global_addr hi (bits 56:32)
          | 0x80000000u;                        // type=2 ("image")
  i32x8 g1;
  g1[0] = (int)((1u << 16)                      // data_size=1 (2 bytes)
                | (1u << 20)                    // pad_enable
                | (3u << 22)                    // pad_interval: 16 DWORDs
                | (3u << 25));                  // pad_amount: 4 DWORDs
  g1[1] = (int)((tensor_d0 & 0xffffu) << 16);   // tensor_dim0[15:0]
  g1[2] = (int)((tensor_d0 >> 16) | ((tensor_d1 & 0xffffu) << 16));
  g1[3] = (int)((tensor_d1 >> 16) | (tile_d0 << 16));
  g1[4] = (int)(tile_d1 & 0xffffu);             // tile_dim1; tile_dim2=0
  g1[5] = (int)stride0;                         // tensor_dim0_stride lo
  g1[6] = 0;
  g1[7] = 0;
  i32x4 z4 = {0, 0, 0, 0};
  i32x8 z8 = {0, 0, 0, 0, 0, 0, 0, 0};
  __builtin_amdgcn_tensor_load_to_lds(g0, g1, z4, z4, z8, 0);
}

// ---------------------------------------------------------------------------
// Prep: x fp32 -> bf16 (row-major unchanged)
// ---------------------------------------------------------------------------
__global__ __launch_bounds__(256) void k_cvt_x(const float* __restrict__ x,
                                               __bf16* __restrict__ xb) {
  size_t i4 = (size_t)blockIdx.x * 256 + threadIdx.x;  // 4 elements each
  if (i4 >= (size_t)MROWS * DM / 4) return;
  float4 f = ((const float4*)x)[i4];
  union { __bf16 h[4]; unsigned long long u; } pk;
  pk.h[0] = (__bf16)f.x;
  pk.h[1] = (__bf16)f.y;
  pk.h[2] = (__bf16)f.z;
  pk.h[3] = (__bf16)f.w;
  *(unsigned long long*)(xb + i4 * 4) = pk.u;
}

// Prep: W (448x448 fp32, [K][N]) -> Wt bf16 [N][K]
__global__ __launch_bounds__(256) void k_cvt_weights(
    const float* __restrict__ Wq, const float* __restrict__ Wk,
    const float* __restrict__ Wv, const float* __restrict__ Wo,
    __bf16* __restrict__ Wtq, __bf16* __restrict__ Wtk,
    __bf16* __restrict__ Wtv, __bf16* __restrict__ Wto) {
  int idx = blockIdx.x * 256 + threadIdx.x;
  if (idx >= DM * DM) return;
  int n = idx / DM, k = idx % DM;
  size_t src = (size_t)k * DM + n;
  Wtq[idx] = (__bf16)Wq[src];
  Wtk[idx] = (__bf16)Wk[src];
  Wtv[idx] = (__bf16)Wv[src];
  Wto[idx] = (__bf16)Wo[src];
}

// SSM diagonal blocks -> ssmt[sel][blk][j][i] = M[blk*32+i][blk*32+j]  (bf16)
__global__ __launch_bounds__(256) void k_cvt_ssm(
    const float* __restrict__ A, const float* __restrict__ B,
    const float* __restrict__ C, __bf16* __restrict__ ssmt) {
  int idx = blockIdx.x * 256 + threadIdx.x;
  if (idx >= 3 * 14 * 32 * 32) return;
  int sel = idx / (14 * 1024);
  int rem = idx % (14 * 1024);
  int blk = rem / 1024;
  int j = (rem % 1024) / 32;
  int i = rem % 32;
  const float* M = (sel == 0) ? A : (sel == 1) ? B : C;
  ssmt[idx] = (__bf16)M[(size_t)(blk * 32 + i) * DM + blk * 32 + j];
}

// ---------------------------------------------------------------------------
// GEMM: out_bf16[M][N] = A_bf16[M][K] * Wt_bf16[N][K]
// Workgroup tile 256x64; 8 waves stacked in M, wave tile 32x64 (8 WMMA/K-step).
// Double-buffered LDS tiles; wave 0 prefetches K-step k+1 via the Tensor Data
// Mover while all waves compute K-step k, then waits TENSORcnt<=2 (in-order
// completion => the older buffer's two DMAs are done).
// ---------------------------------------------------------------------------
__global__ __launch_bounds__(256) void k_qkv_gemm(
    const __bf16* __restrict__ xb,
    const __bf16* __restrict__ Wtq, const __bf16* __restrict__ Wtk,
    const __bf16* __restrict__ Wtv,
    __bf16* __restrict__ qo, __bf16* __restrict__ ko, __bf16* __restrict__ vo) {
  const __bf16* Wt = (blockIdx.z == 0) ? Wtq : (blockIdx.z == 1) ? Wtk : Wtv;
  __bf16* out = (blockIdx.z == 0) ? qo : (blockIdx.z == 1) ? ko : vo;

  __shared__ __align__(16) __bf16 As[2][256][40];
  __shared__ __align__(16) __bf16 Bs[2][64][40];

  const int tid = threadIdx.x;
  const int lane = tid & 31;
  const int half = lane >> 4, l16 = lane & 15;
  const int wv = __builtin_amdgcn_readfirstlane(tid >> 5);  // scalar wave id
  const int wm = wv * 32;
  const int m0 = blockIdx.y * 256, n0 = blockIdx.x * 64;

  v8f acc[2][4];
#pragma unroll
  for (int i = 0; i < 2; ++i)
#pragma unroll
    for (int j = 0; j < 4; ++j) acc[i][j] = v8f_zero();

  const unsigned ldsA0 = lds_off(&As[0][0][0]), ldsA1 = lds_off(&As[1][0][0]);
  const unsigned ldsB0 = lds_off(&Bs[0][0][0]), ldsB1 = lds_off(&Bs[1][0][0]);
  const __bf16* Ag = xb + (size_t)m0 * DM;
  const __bf16* Bg = Wt + (size_t)n0 * DM;

  if (wv == 0) {  // prologue: fill buffer 0 for kt=0
    tdm_load_2d(ldsA0, Ag, DM, MROWS - m0, 32, 256, DM);
    tdm_load_2d(ldsB0, Bg, DM, DM - n0, 32, 64, DM);
  }

  int cur = 0;
  for (int kt = 0; kt < DM; kt += 32) {
    if (wv == 0) {
      if (kt + 32 < DM) {  // prefetch next K-step into the other buffer
        const int kn = kt + 32;
        tdm_load_2d(cur ? ldsA0 : ldsA1, Ag + kn, DM - kn, MROWS - m0, 32, 256,
                    DM);
        tdm_load_2d(cur ? ldsB0 : ldsB1, Bg + kn, DM - kn, DM - n0, 32, 64, DM);
        __builtin_amdgcn_s_wait_tensorcnt(2);  // current buffer complete
      } else {
        __builtin_amdgcn_s_wait_tensorcnt(0);
      }
    }
    __syncthreads();  // current buffer visible to all waves
    v16bf a0 = frag_a(&As[cur][wm + l16][0], half);
    v16bf a1 = frag_a(&As[cur][wm + 16 + l16][0], half);
#pragma unroll
    for (int j = 0; j < 4; ++j) {
      v16bf b = frag_b(&Bs[cur][16 * j + l16][0], half);
      acc[0][j] = wmma_bf16(a0, b, acc[0][j]);
      acc[1][j] = wmma_bf16(a1, b, acc[1][j]);
    }
    __syncthreads();  // reads done before this buffer is refilled
    cur ^= 1;
  }

#pragma unroll
  for (int i = 0; i < 2; ++i)
#pragma unroll
    for (int j = 0; j < 4; ++j) {
      const int col = n0 + 16 * j + l16;
#pragma unroll
      for (int r = 0; r < 8; ++r) {
        const int row = m0 + wm + 16 * i + r + 8 * half;
        out[(size_t)row * DM + col] = (__bf16)acc[i][j][r];
      }
    }
}

// Output GEMM: out fp32 = ssm_bf16 * Wto ; same double-buffered TDM tiling.
__global__ __launch_bounds__(256) void k_out_gemm(
    const __bf16* __restrict__ Abf, const __bf16* __restrict__ Wto,
    float* __restrict__ out) {
  __shared__ __align__(16) __bf16 As[2][256][40];
  __shared__ __align__(16) __bf16 Bs[2][64][40];

  const int tid = threadIdx.x;
  const int lane = tid & 31;
  const int half = lane >> 4, l16 = lane & 15;
  const int wv = __builtin_amdgcn_readfirstlane(tid >> 5);
  const int wm = wv * 32;
  const int m0 = blockIdx.y * 256, n0 = blockIdx.x * 64;

  v8f acc[2][4];
#pragma unroll
  for (int i = 0; i < 2; ++i)
#pragma unroll
    for (int j = 0; j < 4; ++j) acc[i][j] = v8f_zero();

  const unsigned ldsA0 = lds_off(&As[0][0][0]), ldsA1 = lds_off(&As[1][0][0]);
  const unsigned ldsB0 = lds_off(&Bs[0][0][0]), ldsB1 = lds_off(&Bs[1][0][0]);
  const __bf16* Ag = Abf + (size_t)m0 * DM;
  const __bf16* Bg = Wto + (size_t)n0 * DM;

  if (wv == 0) {
    tdm_load_2d(ldsA0, Ag, DM, MROWS - m0, 32, 256, DM);
    tdm_load_2d(ldsB0, Bg, DM, DM - n0, 32, 64, DM);
  }

  int cur = 0;
  for (int kt = 0; kt < DM; kt += 32) {
    if (wv == 0) {
      if (kt + 32 < DM) {
        const int kn = kt + 32;
        tdm_load_2d(cur ? ldsA0 : ldsA1, Ag + kn, DM - kn, MROWS - m0, 32, 256,
                    DM);
        tdm_load_2d(cur ? ldsB0 : ldsB1, Bg + kn, DM - kn, DM - n0, 32, 64, DM);
        __builtin_amdgcn_s_wait_tensorcnt(2);
      } else {
        __builtin_amdgcn_s_wait_tensorcnt(0);
      }
    }
    __syncthreads();
    v16bf a0 = frag_a(&As[cur][wm + l16][0], half);
    v16bf a1 = frag_a(&As[cur][wm + 16 + l16][0], half);
#pragma unroll
    for (int j = 0; j < 4; ++j) {
      v16bf b = frag_b(&Bs[cur][16 * j + l16][0], half);
      acc[0][j] = wmma_bf16(a0, b, acc[0][j]);
      acc[1][j] = wmma_bf16(a1, b, acc[1][j]);
    }
    __syncthreads();
    cur ^= 1;
  }

#pragma unroll
  for (int i = 0; i < 2; ++i)
#pragma unroll
    for (int j = 0; j < 4; ++j) {
      const int col = n0 + 16 * j + l16;
#pragma unroll
      for (int r = 0; r < 8; ++r) {
        const int row = m0 + wm + 16 * i + r + 8 * half;
        out[(size_t)row * DM + col] = acc[i][j][r];
      }
    }
}

// ---------------------------------------------------------------------------
// Fused block-diag (linear + quadratic) + 2x LayerNorm + blend + residual.
// One wave per 16 rows; accumulators cover full D=448 (28 n-tiles).
// ---------------------------------------------------------------------------
__global__ __launch_bounds__(128, 1) void k_ssm_fuse(
    const __bf16* __restrict__ qb, const __bf16* __restrict__ kb,
    const __bf16* __restrict__ vb, const __bf16* __restrict__ ssmt,
    const float* __restrict__ x, const float* __restrict__ dwp,
    const float* __restrict__ gamma, const float* __restrict__ beta,
    __bf16* __restrict__ so) {
  __shared__ float gs[DM], bs[DM];
  const int tid = threadIdx.x;
  for (int i = tid; i < DM; i += 128) {
    gs[i] = gamma[i];
    bs[i] = beta[i];
  }
  __syncthreads();

  const int lane = tid & 31, wave = tid >> 5;
  const int half = lane >> 4, l16 = lane & 15;
  const int rowbase = blockIdx.x * 64 + wave * 16;
  const float dw = 1.f / (1.f + __expf(-dwp[0]));  // sigmoid

  v8f lin[28], quad[28];
#pragma unroll
  for (int t = 0; t < 28; ++t) {
    lin[t] = v8f_zero();
    quad[t] = v8f_zero();
  }

  const __bf16* qrow = qb + (size_t)(rowbase + l16) * DM;
  const __bf16* krow = kb + (size_t)(rowbase + l16) * DM;
  const __bf16* vrow = vb + (size_t)(rowbase + l16) * DM;
  const v8f vz = v8f_zero();

#pragma unroll
  for (int blk = 0; blk < 14; ++blk) {
    const int ko = blk * 32;
    v16bf qf = frag_a(qrow + ko, half);
    v16bf kf = frag_a(krow + ko, half);
    v16bf vf = frag_a(vrow + ko, half);
    // 0.1 * rowsum of the 32-wide block (each lane holds half the K range)
    float sq = frag_sum(qf); sq = (sq + __shfl_xor(sq, 16, 32)) * 0.1f;
    float sk = frag_sum(kf); sk = (sk + __shfl_xor(sk, 16, 32)) * 0.1f;
    float sv = frag_sum(vf); sv = (sv + __shfl_xor(sv, 16, 32)) * 0.1f;
#pragma unroll
    for (int j = 0; j < 2; ++j) {
      const int t = blk * 2 + j;
      const __bf16* bA = ssmt + ((size_t)(0 * 14 + blk) * 32 + 16 * j + l16) * 32;
      const __bf16* bB = ssmt + ((size_t)(1 * 14 + blk) * 32 + 16 * j + l16) * 32;
      const __bf16* bC = ssmt + ((size_t)(2 * 14 + blk) * 32 + 16 * j + l16) * 32;
      v8f y;
      y = wmma_bf16(qf, frag_b(bA, half), vz);
#pragma unroll
      for (int r = 0; r < 8; ++r) {
        float s = __shfl(sq, r + 8 * half, 32);  // scale of output row M=r+8*half
        lin[t][r] += y[r];
        quad[t][r] += y[r] * s;
      }
      y = wmma_bf16(kf, frag_b(bB, half), vz);
#pragma unroll
      for (int r = 0; r < 8; ++r) {
        float s = __shfl(sk, r + 8 * half, 32);
        lin[t][r] += y[r];
        quad[t][r] += y[r] * s;
      }
      y = wmma_bf16(vf, frag_b(bC, half), vz);
#pragma unroll
      for (int r = 0; r < 8; ++r) {
        float s = __shfl(sv, r + 8 * half, 32);
        lin[t][r] += y[r];
        quad[t][r] += y[r] * s;
      }
    }
  }

  // Dual LayerNorm (row stats across 28 tiles x 16 lanes of this half),
  // blend, residual, store bf16.
  const float inv448 = 1.0f / 448.0f;
#pragma unroll
  for (int r = 0; r < 8; ++r) {
    float s1 = 0.f, s2 = 0.f, t1 = 0.f, t2 = 0.f;
#pragma unroll
    for (int t = 0; t < 28; ++t) {
      float a = lin[t][r];
      s1 += a; s2 += a * a;
      float b = quad[t][r];
      t1 += b; t2 += b * b;
    }
#pragma unroll
    for (int m = 1; m <= 8; m <<= 1) {  // reduce within 16-lane half
      s1 += __shfl_xor(s1, m, 32);
      s2 += __shfl_xor(s2, m, 32);
      t1 += __shfl_xor(t1, m, 32);
      t2 += __shfl_xor(t2, m, 32);
    }
    const float muL = s1 * inv448;
    const float rsL = rsqrtf(fmaxf(s2 * inv448 - muL * muL, 0.f) + 1e-5f);
    const float muQ = t1 * inv448;
    const float rsQ = rsqrtf(fmaxf(t2 * inv448 - muQ * muQ, 0.f) + 1e-5f);
    const int row = rowbase + r + 8 * half;
    const float* xr = x + (size_t)row * DM;
    __bf16* orow = so + (size_t)row * DM;
#pragma unroll
    for (int t = 0; t < 28; ++t) {
      const int n = t * 16 + l16;
      const float g = gs[n], bb = bs[n];
      const float eL = (lin[t][r] - muL) * rsL * g + bb;
      const float eQ = (quad[t][r] - muQ) * rsQ * g + bb;
      orow[n] = (__bf16)(dw * eQ + (1.f - dw) * eL + xr[n]);
    }
  }
}

// ---------------------------------------------------------------------------
extern "C" void kernel_launch(void* const* d_in, const int* in_sizes, int n_in,
                              void* d_out, int out_size, void* d_ws,
                              size_t ws_size, hipStream_t stream) {
  const float* x = (const float*)d_in[0];
  const float* Wq = (const float*)d_in[1];
  const float* Wk = (const float*)d_in[2];
  const float* Wv = (const float*)d_in[3];
  const float* Wo = (const float*)d_in[4];
  const float* SA = (const float*)d_in[5];
  const float* SB = (const float*)d_in[6];
  const float* SC = (const float*)d_in[7];
  const float* dwp = (const float*)d_in[8];
  const float* gamma = (const float*)d_in[9];
  const float* beta = (const float*)d_in[10];
  float* out = (float*)d_out;

  // Workspace layout (~295 MB total)
  size_t off = 0;
  auto alloc = [&](size_t bytes) {
    void* p = (char*)d_ws + off;
    off += (bytes + 255) & ~(size_t)255;
    return p;
  };
  __bf16* Wtq = (__bf16*)alloc((size_t)DM * DM * 2);
  __bf16* Wtk = (__bf16*)alloc((size_t)DM * DM * 2);
  __bf16* Wtv = (__bf16*)alloc((size_t)DM * DM * 2);
  __bf16* Wto = (__bf16*)alloc((size_t)DM * DM * 2);
  __bf16* ssmt = (__bf16*)alloc((size_t)3 * 14 * 32 * 32 * 2);
  __bf16* xb = (__bf16*)alloc((size_t)MROWS * DM * 2);
  __bf16* qb = (__bf16*)alloc((size_t)MROWS * DM * 2);
  __bf16* kb = (__bf16*)alloc((size_t)MROWS * DM * 2);
  __bf16* vb = (__bf16*)alloc((size_t)MROWS * DM * 2);
  __bf16* sb = (__bf16*)alloc((size_t)MROWS * DM * 2);

  k_cvt_x<<<(MROWS * DM / 4 + 255) / 256, 256, 0, stream>>>(x, xb);
  k_cvt_weights<<<(DM * DM + 255) / 256, 256, 0, stream>>>(Wq, Wk, Wv, Wo, Wtq,
                                                           Wtk, Wtv, Wto);
  k_cvt_ssm<<<(3 * 14 * 32 * 32 + 255) / 256, 256, 0, stream>>>(SA, SB, SC,
                                                                ssmt);
  k_qkv_gemm<<<dim3(DM / 64, MROWS / 256, 3), 256, 0, stream>>>(
      xb, Wtq, Wtk, Wtv, qb, kb, vb);
  k_ssm_fuse<<<MROWS / 64, 128, 0, stream>>>(qb, kb, vb, ssmt, x, dwp, gamma,
                                             beta, sb);
  k_out_gemm<<<dim3(DM / 64, MROWS / 256), 256, 0, stream>>>(sb, Wto, out);
}
